// ECGCNN_MoE_Large_1005022347833
// MI455X (gfx1250) — compile-verified, matrix operand order checked by reading, not run on hardware
//
#include <hip/hip_runtime.h>
#include <math.h>

typedef __attribute__((ext_vector_type(16))) __bf16 v16bf;
typedef __attribute__((ext_vector_type(8)))  float  v8f;
typedef __attribute__((ext_vector_type(8)))  unsigned int v8u;

// ---------- bf16 helpers (RNE) ----------
__device__ __forceinline__ unsigned short f2bf(float f) {
  union { float f; unsigned u; } v; v.f = f;
  unsigned r = v.u + 0x7FFFu + ((v.u >> 16) & 1u);
  return (unsigned short)(r >> 16);
}
__device__ __forceinline__ float bf2f(unsigned short h) {
  union { unsigned u; float f; } v; v.u = ((unsigned)h) << 16; return v.f;
}

// =====================================================================
// conv1: x (B,1,1024) -> h bf16 (B,16,1024) with ReLU, + pooled mean (B,16)
// =====================================================================
__global__ __launch_bounds__(256) void conv1_kernel(
    const float* __restrict__ x, const float* __restrict__ w,
    const float* __restrict__ bias, unsigned short* __restrict__ h,
    float* __restrict__ pooled) {
  __shared__ float xs[1026];
  __shared__ float csum[16];
  const int b = blockIdx.x, t = threadIdx.x;
  for (int i = t; i < 1026; i += 256) {
    int l = i - 1;
    xs[i] = (l >= 0 && l < 1024) ? x[b * 1024 + l] : 0.f;
  }
  if (t < 16) csum[t] = 0.f;
  __syncthreads();
  const int c = t >> 4, sub = t & 15;
  const float w0 = w[c * 3], w1 = w[c * 3 + 1], w2 = w[c * 3 + 2], bb = bias[c];
  float psum = 0.f;
  for (int j = 0; j < 64; ++j) {
    int l = sub * 64 + j;
    float v = w0 * xs[l] + w1 * xs[l + 1] + w2 * xs[l + 2] + bb;
    v = fmaxf(v, 0.f);
    h[(size_t)(b * 16 + c) * 1024 + l] = f2bf(v);
    psum += v;
  }
  atomicAdd(&csum[c], psum);
  __syncthreads();
  if (t < 16) pooled[b * 16 + t] = csum[t] * (1.f / 1024.f);
}

// =====================================================================
// router: pooled (B,16) -> combine (B,4) top-2 gates; cv^2 -> cv_out
// =====================================================================
__global__ __launch_bounds__(128) void router_kernel(
    const float* __restrict__ pooled, const float* __restrict__ rw,
    const float* __restrict__ rb, float* __restrict__ combine,
    float* __restrict__ cv_out) {
  __shared__ float sum_p[4];
  const int t = threadIdx.x;
  if (t < 4) sum_p[t] = 0.f;
  __syncthreads();
  float p[4]; float mx = -1e30f;
  for (int e = 0; e < 4; ++e) {
    float z = rb[e];
    for (int c = 0; c < 16; ++c) z += pooled[t * 16 + c] * rw[c * 4 + e];
    p[e] = z; mx = fmaxf(mx, z);
  }
  float s = 0.f;
  for (int e = 0; e < 4; ++e) { p[e] = __expf(p[e] - mx); s += p[e]; }
  for (int e = 0; e < 4; ++e) p[e] /= s;
  for (int e = 0; e < 4; ++e) atomicAdd(&sum_p[e], p[e]);
  int i0 = 0;
  for (int e = 1; e < 4; ++e) if (p[e] > p[i0]) i0 = e;
  int i1 = -1;
  for (int e = 0; e < 4; ++e) if (e != i0 && (i1 < 0 || p[e] > p[i1])) i1 = e;
  const float gs = p[i0] + p[i1];
  float cb[4] = {0.f, 0.f, 0.f, 0.f};
  cb[i0] = p[i0] / gs; cb[i1] = p[i1] / gs;
  for (int e = 0; e < 4; ++e) combine[t * 4 + e] = cb[e];
  __syncthreads();
  if (t == 0) {
    float mp[4], mean = 0.f;
    for (int e = 0; e < 4; ++e) { mp[e] = sum_p[e] * (1.f / 128.f); mean += mp[e]; }
    mean *= 0.25f;
    float var = 0.f;
    for (int e = 0; e < 4; ++e) { float d = mp[e] - mean; var += d * d; }
    float sd = sqrtf(var / 3.f);               // ddof=1
    float cv = sd / (mean + 1e-10f);
    *cv_out = cv * cv;
  }
}

// =====================================================================
// WMMA conv1d k=3 pad=1: in bf16 (B,Cin,L) -> out bf16 (B,Cout,L)
// out = conv + bias, optional ReLU.
// Block = 8 waves, tile 64 O x 64 L; each wave owns TWO 16x16 (O x L)
// tiles sharing one B fragment (12 LDS loads per WMMA instead of 16).
// A = weights [tap][o][i] (LDS), B = shifted input slab [l][i] (LDS).
// =====================================================================
__global__ __launch_bounds__(256) void conv_wmma_kernel(
    const unsigned short* __restrict__ in, const float* __restrict__ w,
    const float* __restrict__ bias, unsigned short* __restrict__ out,
    int Cin, int Cout, int Lin, int do_relu) {
  __shared__ unsigned short sA[3 * 64 * 32];   // [k][o][i]
  __shared__ unsigned short sB[66 * 32];       // [l_local][i]
  const int tid  = threadIdx.x;
  const int lane = tid & 31;
  const int wv   = tid >> 5;
  const int lt   = wv & 3;                     // L subtile 0..3
  const int oh   = wv >> 2;                    // O half 0..1 (2 subtiles each)
  const int l0 = blockIdx.x * 64;
  const int o0 = blockIdx.y * 64;
  const int b  = blockIdx.z;
  const unsigned short* inB = in + (size_t)b * Cin * Lin;
  const int m  = lane & 15;
  const int hi = lane >> 4;
  v8f acc0 = {};
  v8f acc1 = {};

  for (int ic0 = 0; ic0 < Cin; ic0 += 32) {
    __syncthreads();
    // stage weights (fp32 -> bf16): sA[k*2048 + o*32 + i]
    for (int idx = tid; idx < 3 * 64 * 32; idx += 256) {
      int k = idx >> 11, rem = idx & 2047, o = rem >> 5, i = rem & 31;
      int oc = o0 + o, ic = ic0 + i;
      float v = (oc < Cout && ic < Cin) ? w[((size_t)oc * Cin + ic) * 3 + k] : 0.f;
      sA[idx] = f2bf(v);
    }
    // stage transposed input slab with halo: sB[ll*32 + i] = in[ic0+i][l0-1+ll]
    for (int idx = tid; idx < 66 * 32; idx += 256) {
      int i = idx / 66, ll = idx % 66;
      int l = l0 - 1 + ll, ic = ic0 + i;
      unsigned short v = 0;
      if (l >= 0 && l < Lin && ic < Cin) v = inB[(size_t)ic * Lin + l];
      sB[ll * 32 + i] = v;
    }
    // prefetch next K-chunk's input rows into cache (global_prefetch_b8)
    {
      int icn = ic0 + 32 + (tid & 31);
      if (icn < Cin)
        __builtin_prefetch(&inB[(size_t)icn * Lin + l0 + ((tid >> 5) << 3)], 0, 1);
    }
    __syncthreads();
#pragma unroll
    for (int k = 0; k < 3; ++k) {
      v8u bu, au0, au1;
#pragma unroll
      for (int v = 0; v < 8; ++v) {
        const int kb = ((v & 4) ? 16 : 0) + ((v & 3) * 2) + hi * 8;
        bu[v]  = *(const unsigned*)&sB[(lt * 16 + m + k) * 32 + kb];
        au0[v] = *(const unsigned*)&sA[k * 2048 + (oh * 32 + m) * 32 + kb];
        au1[v] = *(const unsigned*)&sA[k * 2048 + (oh * 32 + 16 + m) * 32 + kb];
      }
      v16bf Bf = __builtin_bit_cast(v16bf, bu);
      acc0 = __builtin_amdgcn_wmma_f32_16x16x32_bf16(
          false, __builtin_bit_cast(v16bf, au0), false, Bf, (short)0, acc0,
          false, false);
      acc1 = __builtin_amdgcn_wmma_f32_16x16x32_bf16(
          false, __builtin_bit_cast(v16bf, au1), false, Bf, (short)0, acc1,
          false, false);
    }
  }
  // epilogue: two O-subtiles
#pragma unroll
  for (int s = 0; s < 2; ++s) {
    const int obase = o0 + oh * 32 + s * 16;
    if (obase >= Cout) continue;
    unsigned short* outB =
        out + ((size_t)b * Cout + obase) * Lin + l0 + lt * 16;
#pragma unroll
    for (int r = 0; r < 8; ++r) {
      const int M = r + hi * 8;
      float v = (s == 0 ? acc0[r] : acc1[r]) + bias[obase + M];
      if (do_relu) v = fmaxf(v, 0.f);
      outB[(size_t)M * Lin + m] = f2bf(v);
    }
  }
}

// =====================================================================
// maxpool2 along L: (rows, 2*Lout) bf16 -> (rows, Lout) bf16
// =====================================================================
__global__ __launch_bounds__(256) void pool2_kernel(
    const unsigned short* __restrict__ in, unsigned short* __restrict__ out,
    int rows, int Lout) {
  const int idx = blockIdx.x * 256 + threadIdx.x;
  if (idx >= rows * Lout) return;
  const int r = idx / Lout, l = idx % Lout;
  const float a = bf2f(in[(size_t)r * 2 * Lout + 2 * l]);
  const float b = bf2f(in[(size_t)r * 2 * Lout + 2 * l + 1]);
  out[idx] = f2bf(fmaxf(a, b));
}

// =====================================================================
// final pool + gate-combine: in bf16 (B,1024,256) -> out fp32 (B,1024,128)
// out = (e==0 ? 0 : out) + gate[b,e] * maxpool2(in)
// =====================================================================
__global__ __launch_bounds__(256) void pool3accum_kernel(
    const unsigned short* __restrict__ in, const float* __restrict__ combine,
    float* __restrict__ out, int e) {
  const int idx = blockIdx.x * 256 + threadIdx.x;
  if (idx >= 128 * 1024 * 128) return;
  const int l  = idx & 127;
  const int rc = idx >> 7;          // b*1024 + c
  const int b  = rc >> 10;
  const float g = combine[b * 4 + e];
  const float a = bf2f(in[(size_t)rc * 256 + 2 * l]);
  const float c = bf2f(in[(size_t)rc * 256 + 2 * l + 1]);
  const float v = g * fmaxf(a, c);
  if (e == 0) out[idx] = v; else out[idx] += v;
}

// fp32 -> bf16 flatten for fc1 A-matrix
__global__ __launch_bounds__(256) void flatcvt_kernel(
    const float* __restrict__ in, unsigned short* __restrict__ out, int n) {
  const int idx = blockIdx.x * 256 + threadIdx.x;
  if (idx < n) out[idx] = f2bf(in[idx]);
}

__global__ __launch_bounds__(256) void zero_kernel(float* __restrict__ p, int n) {
  const int idx = blockIdx.x * 256 + threadIdx.x;
  if (idx < n) p[idx] = 0.f;
}

// =====================================================================
// fc1 GEMM: A bf16 (128,131072) x W fp32->bf16 (131072,256), split-K,
// fp32 atomic accumulation. 1 wave per block, 16x16 tile, K slice 1024.
// =====================================================================
__global__ __launch_bounds__(32) void fc1_gemm_kernel(
    const unsigned short* __restrict__ Aflat, const float* __restrict__ Wfc,
    float* __restrict__ accb) {
  __shared__ unsigned short sBT[16 * 32];   // [n][k]
  const int lane = threadIdx.x;
  const int m0 = blockIdx.x * 16;
  const int n0 = blockIdx.y * 16;
  const int kbase = blockIdx.z * 1024;
  const int m = lane & 15, hi = lane >> 4;
  v8f c = {};
  for (int kc = 0; kc < 1024; kc += 32) {
    const int k0 = kbase + kc;
    __syncthreads();
    for (int idx = lane; idx < 512; idx += 32) {
      const int k = idx >> 4, n = idx & 15;
      sBT[n * 32 + k] = f2bf(Wfc[(size_t)(k0 + k) * 256 + n0 + n]);
    }
    __syncthreads();
    v8u au, bu;
    const unsigned short* arow = Aflat + (size_t)(m0 + m) * 131072 + k0;
#pragma unroll
    for (int v = 0; v < 8; ++v) {
      const int kb = ((v & 4) ? 16 : 0) + ((v & 3) * 2) + hi * 8;
      au[v] = *(const unsigned*)&arow[kb];
      bu[v] = *(const unsigned*)&sBT[m * 32 + kb];
    }
    v16bf A  = __builtin_bit_cast(v16bf, au);
    v16bf Bf = __builtin_bit_cast(v16bf, bu);
    c = __builtin_amdgcn_wmma_f32_16x16x32_bf16(false, A, false, Bf,
                                                (short)0, c, false, false);
  }
#pragma unroll
  for (int r = 0; r < 8; ++r)
    atomicAdd(&accb[(size_t)(m0 + r + hi * 8) * 256 + n0 + m], c[r]);
}

__global__ __launch_bounds__(256) void fc1epi_kernel(
    const float* __restrict__ acc, const float* __restrict__ b1,
    float* __restrict__ z1) {
  const int idx = blockIdx.x * 256 + threadIdx.x;
  if (idx < 128 * 256) z1[idx] = fmaxf(acc[idx] + b1[idx & 255], 0.f);
}

// =====================================================================
// fc2 (256->64) + ReLU + fc3 (64->5): one block of 64 threads per sample
// =====================================================================
__global__ __launch_bounds__(64) void fc23_kernel(
    const float* __restrict__ z1, const float* __restrict__ w2,
    const float* __restrict__ b2, const float* __restrict__ w3,
    const float* __restrict__ b3, float* __restrict__ y) {
  __shared__ float s1[256];
  __shared__ float s2[64];
  const int b = blockIdx.x, t = threadIdx.x;
  for (int i = t; i < 256; i += 64) s1[i] = z1[b * 256 + i];
  __syncthreads();
  float z = b2[t];
  for (int k = 0; k < 256; ++k) z += s1[k] * w2[k * 64 + t];
  s2[t] = fmaxf(z, 0.f);
  __syncthreads();
  if (t < 5) {
    float o = b3[t];
    for (int k = 0; k < 64; ++k) o += s2[k] * w3[k * 5 + t];
    y[b * 5 + t] = o;
  }
}

// =====================================================================
extern "C" void kernel_launch(void* const* d_in, const int* in_sizes, int n_in,
                              void* d_out, int out_size, void* d_ws, size_t ws_size,
                              hipStream_t stream) {
  const float* x        = (const float*)d_in[0];
  const float* conv1_w  = (const float*)d_in[1];
  const float* conv1_b  = (const float*)d_in[2];
  const float* ew[6] = {(const float*)d_in[3],  (const float*)d_in[5],
                        (const float*)d_in[7],  (const float*)d_in[9],
                        (const float*)d_in[11], (const float*)d_in[13]};
  const float* eb[6] = {(const float*)d_in[4],  (const float*)d_in[6],
                        (const float*)d_in[8],  (const float*)d_in[10],
                        (const float*)d_in[12], (const float*)d_in[14]};
  const float* router_w = (const float*)d_in[15];
  const float* router_b = (const float*)d_in[16];
  const float* fc1_w    = (const float*)d_in[17];
  const float* fc1_b    = (const float*)d_in[18];
  const float* fc2_w    = (const float*)d_in[19];
  const float* fc2_b    = (const float*)d_in[20];
  const float* fc3_w    = (const float*)d_in[21];
  const float* fc3_b    = (const float*)d_in[22];
  float* outF = (float*)d_out;      // y (128*5) then cv^2 (1)

  const int CO[6] = {32, 64, 128, 256, 512, 1024};
  const int WSTRIDE[6] = {32*16*3, 64*32*3, 128*64*3, 256*128*3, 512*256*3, 1024*512*3};

  char* wsb = (char*)d_ws;
  const size_t H_OFF      = 0;                          // 4,194,304 B (bf16 h)
  const size_t POOLED_OFF = H_OFF + 4194304;            // 8,192 B
  const size_t COMB_OFF   = POOLED_OFF + 8192;          // 2,048 B
  const size_t BUFA_OFF   = COMB_OFF + 2048;            // 33,554,432 B
  const size_t BUFB_OFF   = BUFA_OFF + 33554432;        // 67,108,864 B
  const size_t OUTACC_OFF = BUFB_OFF + 67108864;        // 67,108,864 B
  const size_t FC1A_OFF   = OUTACC_OFF + 67108864;      // 131,072 B
  const size_t Z1_OFF     = FC1A_OFF + 131072;          // 131,072 B

  unsigned short* h      = (unsigned short*)(wsb + H_OFF);
  float*          pooled = (float*)(wsb + POOLED_OFF);
  float*          comb   = (float*)(wsb + COMB_OFF);
  unsigned short* bufA   = (unsigned short*)(wsb + BUFA_OFF);
  unsigned short* bufB   = (unsigned short*)(wsb + BUFB_OFF);
  float*          outAcc = (float*)(wsb + OUTACC_OFF);
  unsigned short* flatbf = bufA;   // reused after expert loop
  float*          fc1acc = (float*)(wsb + FC1A_OFF);
  float*          z1     = (float*)(wsb + Z1_OFF);

  conv1_kernel<<<128, 256, 0, stream>>>(x, conv1_w, conv1_b, h, pooled);
  router_kernel<<<1, 128, 0, stream>>>(pooled, router_w, router_b, comb, outF + 640);

  for (int e = 0; e < 4; ++e) {
    const float* w1 = ew[0] + (size_t)e * WSTRIDE[0]; const float* b1 = eb[0] + e * CO[0];
    const float* w2 = ew[1] + (size_t)e * WSTRIDE[1]; const float* b2 = eb[1] + e * CO[1];
    const float* w3 = ew[2] + (size_t)e * WSTRIDE[2]; const float* b3 = eb[2] + e * CO[2];
    const float* w4 = ew[3] + (size_t)e * WSTRIDE[3]; const float* b4 = eb[3] + e * CO[3];
    const float* w5 = ew[4] + (size_t)e * WSTRIDE[4]; const float* b5 = eb[4] + e * CO[4];
    const float* w6 = ew[5] + (size_t)e * WSTRIDE[5]; const float* b6 = eb[5] + e * CO[5];

    // block1: conv(16->32,L1024) conv(32->64)+relu pool -> (64,512)
    conv_wmma_kernel<<<dim3(16, 1, 128), 256, 0, stream>>>(h,    w1, b1, bufA, 16, 32, 1024, 0);
    conv_wmma_kernel<<<dim3(16, 1, 128), 256, 0, stream>>>(bufA, w2, b2, bufB, 32, 64, 1024, 1);
    { int n = 128 * 64 * 512;
      pool2_kernel<<<(n + 255) / 256, 256, 0, stream>>>(bufB, bufA, 128 * 64, 512); }
    // block2: conv(64->128,L512) conv(128->256)+relu pool -> (256,256)
    conv_wmma_kernel<<<dim3(8, 2, 128), 256, 0, stream>>>(bufA, w3, b3, bufB, 64, 128, 512, 0);
    conv_wmma_kernel<<<dim3(8, 4, 128), 256, 0, stream>>>(bufB, w4, b4, bufA, 128, 256, 512, 1);
    { int n = 128 * 256 * 256;
      pool2_kernel<<<(n + 255) / 256, 256, 0, stream>>>(bufA, bufB, 128 * 256, 256); }
    // block3: conv(256->512,L256) conv(512->1024)+relu pool+gate-> outAcc
    conv_wmma_kernel<<<dim3(4, 8, 128), 256, 0, stream>>>(bufB, w5, b5, bufA, 256, 512, 256, 0);
    conv_wmma_kernel<<<dim3(4, 16, 128), 256, 0, stream>>>(bufA, w6, b6, bufB, 512, 1024, 256, 1);
    { int n = 128 * 1024 * 128;
      pool3accum_kernel<<<(n + 255) / 256, 256, 0, stream>>>(bufB, comb, outAcc, e); }
  }

  // fc1: bf16 flatten, zero accum, split-K WMMA GEMM, bias+relu
  { int n = 128 * 131072;
    flatcvt_kernel<<<(n + 255) / 256, 256, 0, stream>>>(outAcc, flatbf, n); }
  { int n = 128 * 256;
    zero_kernel<<<(n + 255) / 256, 256, 0, stream>>>(fc1acc, n); }
  fc1_gemm_kernel<<<dim3(8, 16, 128), 32, 0, stream>>>(flatbf, fc1_w, fc1acc);
  { int n = 128 * 256;
    fc1epi_kernel<<<(n + 255) / 256, 256, 0, stream>>>(fc1acc, fc1_b, z1); }

  fc23_kernel<<<128, 64, 0, stream>>>(z1, fc2_w, fc2_b, fc3_w, fc3_b, outF);

  (void)in_sizes; (void)n_in; (void)out_size; (void)ws_size;
}
